// EGNNWithRBF_50491635531837
// MI455X (gfx1250) — matrix operand order repaired and checked by reading, not compiled, not run on hardware
//
#include <hip/hip_runtime.h>
#include <hip/hip_bf16.h>
#include <cstddef>

// ---------------------------------------------------------------------------
// EGNN forward for MI455X (gfx1250, wave32, WMMA).
// Big edge GEMMs (65536x128 @ 128x128, K=128) on v_wmma_f32_16x16x32_f16.
// Block (8 waves) = 128-row strip x full N=128; weights staged in LDS once per
// block (padded rows -> conflict-free); per wave: all 4 A fragments preloaded
// as one vmem clause, per K-block all 8 B fragments batch-loaded from LDS,
// then 8 independent WMMAs issued back-to-back.
// ---------------------------------------------------------------------------

typedef _Float16 half_t;
typedef __attribute__((ext_vector_type(8)))  _Float16 v8h;
typedef __attribute__((ext_vector_type(16))) _Float16 v16h;
typedef __attribute__((ext_vector_type(8)))  float    v8f;

#define H 128
#define N_ATOMS 256
#define N_EDGES (N_ATOMS * N_ATOMS)
#define NRBF 50
#define TD 64
#define LDS_STRIDE 136   // 128 + 8 halves pad: row stride 272B -> bank = 4n mod 64

__device__ __forceinline__ float silu_f(float v) {
    return v / (1.0f + __expf(-v));
}

// Stage a 128x128 f16 weight matrix (row-major) into padded LDS. 256 threads.
__device__ __forceinline__ void stage_weights(const half_t* __restrict__ W,
                                              half_t* __restrict__ Wlds) {
    for (int v = threadIdx.x; v < (H * H) / 8; v += 256) {   // 2048 x v8h
        const int r  = v >> 4;
        const int c8 = (v & 15) << 3;
        *(v8h*)(&Wlds[r * LDS_STRIDE + c8]) = *(const v8h*)(W + r * H + c8);
    }
    __syncthreads();
}

// One wave: 16-row strip of A times all 8 column tiles (N=128), K=128.
__device__ __forceinline__ void wmma_strip(const half_t* __restrict__ A,
                                           const half_t* __restrict__ Wlds,
                                           int rowBase, v8f acc[8]) {
    const int lane = threadIdx.x & 31;
    const int lrow = lane & 15;
    const int hi   = lane >> 4;
    const half_t* ar = A + (size_t)(rowBase + lrow) * H;

    // Preload all 4 A fragments (8 x global_load_b128, one clause).
    v16h afr[4];
#pragma unroll
    for (int k = 0; k < 4; ++k) {
        const int kb = k * 32;
        v8h alo = *(const v8h*)(ar + kb + 8 * hi);
        v8h ahi = *(const v8h*)(ar + kb + 16 + 8 * hi);
        afr[k] = __builtin_shufflevector(alo, ahi,
                    0,1,2,3,4,5,6,7,8,9,10,11,12,13,14,15);
    }

#pragma unroll
    for (int k = 0; k < 4; ++k) {
        const int kb = k * 32;
        // Batch-load all 8 B fragments for this K-block from LDS.
        v16h bfr[8];
#pragma unroll
        for (int ct = 0; ct < 8; ++ct) {
            const half_t* wl = &Wlds[(size_t)(ct * 16 + lrow) * LDS_STRIDE
                                     + kb + 8 * hi];
            v8h blo = *(const v8h*)(wl);
            v8h bhi = *(const v8h*)(wl + 16);
            bfr[ct] = __builtin_shufflevector(blo, bhi,
                        0,1,2,3,4,5,6,7,8,9,10,11,12,13,14,15);
        }
        // 8 independent WMMAs back-to-back (distinct accumulators: no hazards).
#pragma unroll
        for (int ct = 0; ct < 8; ++ct) {
            acc[ct] = __builtin_amdgcn_wmma_f32_16x16x32_f16(
                          false, afr[k], false, bfr[ct], (short)0, acc[ct],
                          false, false);
        }
    }
}

// ---------------------------------------------------------------------------
// Time embedding + initial node features
// ---------------------------------------------------------------------------
__global__ void init_h_kernel(const float* __restrict__ t,
                              const int* __restrict__ atype,
                              const float* __restrict__ time_w1, const float* __restrict__ time_b1,
                              const float* __restrict__ time_w2, const float* __restrict__ time_b2,
                              const float* __restrict__ atom_emb,
                              const float* __restrict__ node_w, const float* __restrict__ node_b,
                              float* __restrict__ h) {
    __shared__ float temb0[TD];
    __shared__ float temb1[H];
    const int o = threadIdx.x;
    const int i = blockIdx.x;
    const float tv = t[0];
    if (o < TD / 2) {
        float freq = __expf((float)o * (-logf(10000.0f) / (float)(TD / 2 - 1)));
        float ang = tv * freq;
        temb0[o] = __sinf(ang);
        temb0[o + TD / 2] = __cosf(ang);
    }
    __syncthreads();
    float s = time_b1[o];
    for (int k = 0; k < TD; ++k) s += temb0[k] * time_w1[o * TD + k];
    temb1[o] = silu_f(s);
    __syncthreads();
    float s2 = time_b2[o];
    for (int k = 0; k < H; ++k) s2 += temb1[k] * time_w2[o * H + k];
    const int ty = atype[i];
    float hv = node_b[o];
    for (int k = 0; k < H; ++k) hv += atom_emb[ty * H + k] * node_w[o * H + k];
    h[i * H + o] = hv + s2;
}

// edge_attr (f16) : E[e][o] = sum_r rbf(e,r) * eW[o][r] + eb[o]
__global__ void edge_attr_kernel(const float* __restrict__ x,
                                 const float* __restrict__ centers,
                                 const float* __restrict__ width,
                                 const float* __restrict__ eW,
                                 const float* __restrict__ eb,
                                 half_t* __restrict__ E) {
    __shared__ float rbf[NRBF];
    const int j = blockIdx.x, i = blockIdx.y, o = threadIdx.x;
    const float dx = x[i * 3 + 0] - x[j * 3 + 0];
    const float dy = x[i * 3 + 1] - x[j * 3 + 1];
    const float dz = x[i * 3 + 2] - x[j * 3 + 2];
    const float dist = sqrtf(dx * dx + dy * dy + dz * dz);
    if (o < NRBF) {
        float w = width[0];
        float d = dist - centers[o];
        rbf[o] = __expf(-(d * d) / (2.0f * w * w));
    }
    __syncthreads();
    float s = eb[o];
    for (int r = 0; r < NRBF; ++r) s += rbf[r] * eW[o * NRBF + r];
    E[((size_t)(i * N_ATOMS + j)) * H + o] = (half_t)s;
}

// Convert a 128x128 f32 weight block (with row stride / column offset) to f16.
__global__ void conv_w_kernel(const float* __restrict__ src, int stride, int coloff,
                              half_t* __restrict__ dst) {
    const int idx = blockIdx.x * blockDim.x + threadIdx.x;   // 16384
    const int r = idx >> 7, c = idx & 127;
    dst[idx] = (half_t)src[r * stride + coloff + c];
}

// hWi = h @ W_i^T, hWj = h @ W_j^T  (column slices of edge_w1, stride 385)
__global__ void hproj_kernel(const float* __restrict__ h, const float* __restrict__ eW1,
                             float* __restrict__ hWi, float* __restrict__ hWj) {
    const int i = blockIdx.x, o = threadIdx.x;
    const float* wr = eW1 + o * 385;
    float si = 0.f, sj = 0.f;
    for (int k = 0; k < H; ++k) {
        float hv = h[i * H + k];
        si += hv * wr[k];
        sj += hv * wr[H + k];
    }
    hWi[i * H + o] = si;
    hWj[i * H + o] = sj;
}

// radial[e] = |x_i - x_j|^2
__global__ void radial_kernel(const float* __restrict__ x, float* __restrict__ radial) {
    const int i = blockIdx.x, j = threadIdx.x;
    const float dx = x[i * 3 + 0] - x[j * 3 + 0];
    const float dy = x[i * 3 + 1] - x[j * 3 + 1];
    const float dz = x[i * 3 + 2] - x[j * 3 + 2];
    radial[i * N_ATOMS + j] = dx * dx + dy * dy + dz * dz;
}

// ---------------------------------------------------------------------------
// G1: pre = E @ W_e^T + hWi[i] + hWj[j] + radial*w_r + b1 ; A1 = silu(pre)
// 512 blocks x 256 threads
// ---------------------------------------------------------------------------
__global__ void __launch_bounds__(256)
edge_pre_kernel(const half_t* __restrict__ E, const half_t* __restrict__ We16,
                const float* __restrict__ hWi, const float* __restrict__ hWj,
                const float* __restrict__ radial, const float* __restrict__ eW1,
                const float* __restrict__ eb1, half_t* __restrict__ A1) {
    __shared__ half_t Wlds[H * LDS_STRIDE];
    stage_weights(We16, Wlds);

    const int wave = threadIdx.x >> 5;
    const int rowBase = blockIdx.x * 128 + wave * 16;       // edge base
    v8f acc[8] = {};
    wmma_strip(E, Wlds, rowBase, acc);

    const int lane = threadIdx.x & 31;
    const int lrow = lane & 15;
    const int hi   = lane >> 4;
    const int i_atom = rowBase >> 8;                        // constant per tile
    float rad_r[8];
    int   j_r[8];
#pragma unroll
    for (int r = 0; r < 8; ++r) {
        const int edge = rowBase + r + 8 * hi;
        j_r[r]   = edge & 255;
        rad_r[r] = radial[edge];
    }
#pragma unroll
    for (int ct = 0; ct < 8; ++ct) {
        const int col = ct * 16 + lrow;
        const float wr_c  = eW1[col * 385 + 2 * H];
        const float b1_c  = eb1[col];
        const float hwi_c = hWi[i_atom * H + col];
#pragma unroll
        for (int r = 0; r < 8; ++r) {
            const int edge = rowBase + r + 8 * hi;
            float v = acc[ct][r] + hwi_c + hWj[j_r[r] * H + col]
                    + rad_r[r] * wr_c + b1_c;
            A1[(size_t)edge * H + col] = (half_t)silu_f(v);
        }
    }
}

// G2/G3: out = silu(A @ W^T + bias) (f16), same strip tiling
__global__ void __launch_bounds__(256)
gemm_bias_silu_kernel(const half_t* __restrict__ A, const half_t* __restrict__ W16,
                      const float* __restrict__ bias, half_t* __restrict__ out) {
    __shared__ half_t Wlds[H * LDS_STRIDE];
    stage_weights(W16, Wlds);

    const int wave = threadIdx.x >> 5;
    const int rowBase = blockIdx.x * 128 + wave * 16;
    v8f acc[8] = {};
    wmma_strip(A, Wlds, rowBase, acc);

    const int lane = threadIdx.x & 31;
    const int lrow = lane & 15;
    const int hi   = lane >> 4;
#pragma unroll
    for (int ct = 0; ct < 8; ++ct) {
        const int col = ct * 16 + lrow;
        const float b_c = bias[col];
#pragma unroll
        for (int r = 0; r < 8; ++r) {
            const int edge = rowBase + r + 8 * hi;
            out[(size_t)edge * H + col] = (half_t)silu_f(acc[ct][r] + b_c);
        }
    }
}

// Fused: att = sigmoid(m . att_w + att_b); m *= att.   block = edge, 128 thr
__global__ void att_scale_kernel(half_t* __restrict__ M,
                                 const float* __restrict__ att_w,
                                 const float* __restrict__ att_b) {
    __shared__ float red[H];
    const int e = blockIdx.x;
    const int t = threadIdx.x;
    half_t* mr = M + (size_t)e * H;
    const float mv = (float)mr[t];
    red[t] = mv * att_w[t];
    __syncthreads();
    for (int s = 64; s > 0; s >>= 1) {
        if (t < s) red[t] += red[t + s];
        __syncthreads();
    }
    const float att = 1.0f / (1.0f + __expf(-(red[0] + att_b[0])));
    mr[t] = (half_t)(mv * att);
}

// m_i[i][o] = sum_j m[(i,j)][o]
__global__ void msum_kernel(const half_t* __restrict__ M, float* __restrict__ m_i) {
    const int i = blockIdx.x, o = threadIdx.x;
    float s = 0.f;
    for (int j = 0; j < N_ATOMS; ++j)
        s += (float)M[((size_t)(i * N_ATOMS + j)) * H + o];
    m_i[i * H + o] = s;
}

// h += silu([h, m_i] @ W1^T + b1) @ W2^T + b2
__global__ void node_mlp_kernel(float* __restrict__ h, const float* __restrict__ m_i,
                                const float* __restrict__ W1, const float* __restrict__ b1,
                                const float* __restrict__ W2, const float* __restrict__ b2) {
    __shared__ float t1[H];
    const int i = blockIdx.x, o = threadIdx.x;
    const float* w = W1 + o * (2 * H);
    float s = b1[o];
    for (int k = 0; k < H; ++k) s += h[i * H + k] * w[k];
    for (int k = 0; k < H; ++k) s += m_i[i * H + k] * w[H + k];
    t1[o] = silu_f(s);
    __syncthreads();
    float s2 = b2[o];
    for (int k = 0; k < H; ++k) s2 += t1[k] * W2[o * H + k];
    h[i * H + o] += s2;
}

// cw[e] = (i==j) ? 0 : c1[e] . coord_w2   (block = edge, coalesced reduce)
__global__ void cw_kernel(const half_t* __restrict__ C1,
                          const float* __restrict__ coord_w2,
                          float* __restrict__ cw) {
    __shared__ float red[H];
    const int e = blockIdx.x;
    const int t = threadIdx.x;
    red[t] = (float)C1[(size_t)e * H + t] * coord_w2[t];
    __syncthreads();
    for (int s = 64; s > 0; s >>= 1) {
        if (t < s) red[t] += red[t + s];
        __syncthreads();
    }
    if (t == 0) {
        const int i = e >> 8, j = e & 255;
        cw[e] = (i == j) ? 0.f : red[0];
    }
}

// x_new[i] = x_old[i] + mean_j (x_old[i]-x_old[j]) * cw[i,j]
__global__ void coord_update_kernel(const float* __restrict__ x_old,
                                    const float* __restrict__ cw,
                                    float* __restrict__ x_new) {
    __shared__ float red[3 * 128];
    const int i = blockIdx.x, t = threadIdx.x;     // 128 threads
    const float xi0 = x_old[i * 3 + 0], xi1 = x_old[i * 3 + 1], xi2 = x_old[i * 3 + 2];
    float a0 = 0.f, a1 = 0.f, a2 = 0.f;
    for (int j = t; j < N_ATOMS; j += 128) {
        const float c = cw[i * N_ATOMS + j];
        a0 += (xi0 - x_old[j * 3 + 0]) * c;
        a1 += (xi1 - x_old[j * 3 + 1]) * c;
        a2 += (xi2 - x_old[j * 3 + 2]) * c;
    }
    red[t] = a0; red[128 + t] = a1; red[256 + t] = a2;
    __syncthreads();
    for (int s = 64; s > 0; s >>= 1) {
        if (t < s) {
            red[t] += red[t + s];
            red[128 + t] += red[128 + t + s];
            red[256 + t] += red[256 + t + s];
        }
        __syncthreads();
    }
    if (t < 3) x_new[i * 3 + t] = x_old[i * 3 + t] + red[t * 128] * (1.0f / (float)N_ATOMS);
}

__global__ void copy_x_kernel(const float* __restrict__ src, float* __restrict__ dst) {
    const int idx = blockIdx.x * blockDim.x + threadIdx.x;
    if (idx < N_ATOMS * 3) dst[idx] = src[idx];
}

// out = (x_final - x_orig) + silu(h @ out_w1^T + b1) @ out_w2^T + b2
__global__ void output_kernel(const float* __restrict__ h, const float* __restrict__ x_fin,
                              const float* __restrict__ x_in,
                              const float* __restrict__ oW1, const float* __restrict__ ob1,
                              const float* __restrict__ oW2, const float* __restrict__ ob2,
                              float* __restrict__ out) {
    __shared__ float t1[H];
    const int i = blockIdx.x, o = threadIdx.x;
    float s = ob1[o];
    for (int k = 0; k < H; ++k) s += h[i * H + k] * oW1[o * H + k];
    t1[o] = silu_f(s);
    __syncthreads();
    if (o < 3) {
        float s2 = ob2[o];
        for (int k = 0; k < H; ++k) s2 += t1[k] * oW2[o * H + k];
        out[i * 3 + o] = (x_fin[i * 3 + o] - x_in[i * 3 + o]) + s2;
    }
}

// ---------------------------------------------------------------------------
// Host launcher
// ---------------------------------------------------------------------------
extern "C" void kernel_launch(void* const* d_in, const int* in_sizes, int n_in,
                              void* d_out, int out_size, void* d_ws, size_t ws_size,
                              hipStream_t stream) {
    // jax pytree order: dicts sorted by key, lists in order:
    // 0:x 1:t 2:atom_types | params: 3:atom_emb 4:edge_emb_b 5:edge_emb_w
    // 6+13*L + {0:att_b 1:att_w 2:coord_b1 3:coord_w1 4:coord_w2 5:edge_b1
    //           6:edge_b2 7:edge_w1 8:edge_w2 9:node_b1 10:node_b2 11:node_w1 12:node_w2}
    // 58:node_b 59:node_w 60:out_b1 61:out_b2 62:out_w1 63:out_w2
    // 64:time_b1 65:time_b2 66:time_w1 67:time_w2 | 68:rbf_centers 69:rbf_width
    const float* x_in       = (const float*)d_in[0];
    const float* t_in       = (const float*)d_in[1];
    const int*   atype      = (const int*)d_in[2];
    const float* atom_emb   = (const float*)d_in[3];
    const float* edge_emb_b = (const float*)d_in[4];
    const float* edge_emb_w = (const float*)d_in[5];
    const float* node_b     = (const float*)d_in[58];
    const float* node_w     = (const float*)d_in[59];
    const float* out_b1     = (const float*)d_in[60];
    const float* out_b2     = (const float*)d_in[61];
    const float* out_w1     = (const float*)d_in[62];
    const float* out_w2     = (const float*)d_in[63];
    const float* time_b1    = (const float*)d_in[64];
    const float* time_b2    = (const float*)d_in[65];
    const float* time_w1    = (const float*)d_in[66];
    const float* time_w2    = (const float*)d_in[67];
    const float* rbf_c      = (const float*)d_in[68];
    const float* rbf_w      = (const float*)d_in[69];

    auto LP = [&](int L, int which) -> const float* {
        return (const float*)d_in[6 + L * 13 + which];
    };

    // Workspace layout
    char* ws = (char*)d_ws;
    const size_t MB16 = (size_t)N_EDGES * H * sizeof(half_t);      // 16 MB
    half_t* E   = (half_t*)(ws);
    half_t* A1  = (half_t*)(ws + MB16);          // silu(pre), reused as c1
    half_t* Mb  = (half_t*)(ws + 2 * MB16);      // m
    size_t off  = 3 * MB16;
    float* radial = (float*)(ws + off); off += (size_t)N_EDGES * 4;
    float* cw     = (float*)(ws + off); off += (size_t)N_EDGES * 4;
    float* h      = (float*)(ws + off); off += (size_t)N_ATOMS * H * 4;
    float* m_i    = (float*)(ws + off); off += (size_t)N_ATOMS * H * 4;
    float* hWi    = (float*)(ws + off); off += (size_t)N_ATOMS * H * 4;
    float* hWj    = (float*)(ws + off); off += (size_t)N_ATOMS * H * 4;
    float* xA     = (float*)(ws + off); off += 1024;
    float* xB     = (float*)(ws + off); off += 1024;
    half_t* wf16  = (half_t*)(ws + off);         // 12 x 128x128 f16

    // 1) x copy + node init + edge attributes
    copy_x_kernel<<<3, 256, 0, stream>>>(x_in, xA);
    init_h_kernel<<<N_ATOMS, H, 0, stream>>>(t_in, atype, time_w1, time_b1,
                                             time_w2, time_b2, atom_emb,
                                             node_w, node_b, h);
    edge_attr_kernel<<<dim3(N_ATOMS, N_ATOMS), H, 0, stream>>>(
        x_in, rbf_c, rbf_w, edge_emb_w, edge_emb_b, E);

    // 2) weight conversion to f16
    for (int L = 0; L < 4; ++L) {
        half_t* We16  = wf16 + (size_t)(3 * L + 0) * H * H;
        half_t* W216  = wf16 + (size_t)(3 * L + 1) * H * H;
        half_t* Wc116 = wf16 + (size_t)(3 * L + 2) * H * H;
        conv_w_kernel<<<64, 256, 0, stream>>>(LP(L, 7), 385, 2 * H + 1, We16);
        conv_w_kernel<<<64, 256, 0, stream>>>(LP(L, 8), H, 0, W216);
        conv_w_kernel<<<64, 256, 0, stream>>>(LP(L, 3), H, 0, Wc116);
    }

    // 3) layers
    float* x_cur = xA;
    float* x_nxt = xB;
    const int GEMM_BLOCKS = N_EDGES / 128;   // 512
    for (int L = 0; L < 4; ++L) {
        half_t* We16  = wf16 + (size_t)(3 * L + 0) * H * H;
        half_t* W216  = wf16 + (size_t)(3 * L + 1) * H * H;
        half_t* Wc116 = wf16 + (size_t)(3 * L + 2) * H * H;

        hproj_kernel<<<N_ATOMS, H, 0, stream>>>(h, LP(L, 7), hWi, hWj);
        radial_kernel<<<N_ATOMS, N_ATOMS, 0, stream>>>(x_cur, radial);

        // G1: pre + fused broadcast/bias/silu -> A1 (f16)
        edge_pre_kernel<<<GEMM_BLOCKS, 256, 0, stream>>>(
            E, We16, hWi, hWj, radial, LP(L, 7), LP(L, 5), A1);
        // G2: m = silu(A1 @ edge_w2^T + b2)
        gemm_bias_silu_kernel<<<GEMM_BLOCKS, 256, 0, stream>>>(A1, W216, LP(L, 6), Mb);

        // attention (fused sigmoid-gate, in place on m)
        att_scale_kernel<<<N_EDGES, H, 0, stream>>>(Mb, LP(L, 1), LP(L, 0));
        msum_kernel<<<N_ATOMS, H, 0, stream>>>(Mb, m_i);
        node_mlp_kernel<<<N_ATOMS, H, 0, stream>>>(h, m_i, LP(L, 11), LP(L, 9),
                                                   LP(L, 12), LP(L, 10));

        // G3: c1 = silu(m @ coord_w1^T + cb1) -> reuse A1 buffer
        gemm_bias_silu_kernel<<<GEMM_BLOCKS, 256, 0, stream>>>(Mb, Wc116, LP(L, 2), A1);
        cw_kernel<<<N_EDGES, H, 0, stream>>>(A1, LP(L, 4), cw);
        coord_update_kernel<<<N_ATOMS, 128, 0, stream>>>(x_cur, cw, x_nxt);

        float* tmp = x_cur; x_cur = x_nxt; x_nxt = tmp;
    }

    // 4) output
    output_kernel<<<N_ATOMS, H, 0, stream>>>(h, x_cur, x_in, out_w1, out_b1,
                                             out_w2, out_b2, (float*)d_out);
}
